// ProbSparseSelfAttention_12472585027708
// MI455X (gfx1250) — matrix operand order repaired and checked by reading, not compiled, not run on hardware
//
#include <hip/hip_runtime.h>
#include <math.h>

typedef __bf16 bf16;
typedef __attribute__((ext_vector_type(16))) __bf16 v16bf;
typedef __attribute__((ext_vector_type(8)))  __bf16 bf16x8;
typedef __attribute__((ext_vector_type(8)))  float  v8f;

#define B_  8
#define L_  4096
#define DM_ 1024
#define H_  16
#define DH_ 64
#define U_  45
#define UP_ 48
#define SCALE_ 0.125f

__device__ __forceinline__ v8f zero8() {
    v8f r;
#pragma unroll
    for (int i = 0; i < 8; i++) r[i] = 0.0f;
    return r;
}
__device__ __forceinline__ bf16x8 cvt8(const float* p) {
    float4 a = *reinterpret_cast<const float4*>(p);
    float4 b = *reinterpret_cast<const float4*>(p + 4);
    bf16x8 r;
    r[0]=(bf16)a.x; r[1]=(bf16)a.y; r[2]=(bf16)a.z; r[3]=(bf16)a.w;
    r[4]=(bf16)b.x; r[5]=(bf16)b.y; r[6]=(bf16)b.z; r[7]=(bf16)b.w;
    return r;
}
__device__ __forceinline__ v16bf cvt16(float4 a0, float4 a1, float4 a2, float4 a3) {
    v16bf r;
    r[0]=(bf16)a0.x;  r[1]=(bf16)a0.y;  r[2]=(bf16)a0.z;  r[3]=(bf16)a0.w;
    r[4]=(bf16)a1.x;  r[5]=(bf16)a1.y;  r[6]=(bf16)a1.z;  r[7]=(bf16)a1.w;
    r[8]=(bf16)a2.x;  r[9]=(bf16)a2.y;  r[10]=(bf16)a2.z; r[11]=(bf16)a2.w;
    r[12]=(bf16)a3.x; r[13]=(bf16)a3.y; r[14]=(bf16)a3.z; r[15]=(bf16)a3.w;
    return r;
}
__device__ __forceinline__ v16bf join16(bf16x8 lo, bf16x8 hi) {
    v16bf r;
#pragma unroll
    for (int i = 0; i < 8; i++) { r[i] = lo[i]; r[i+8] = hi[i]; }
    return r;
}
__device__ __forceinline__ v8f wmma_bf16(v16bf a, v16bf b, v8f c) {
    return __builtin_amdgcn_wmma_f32_16x16x32_bf16(false, a, false, b, (short)0, c, false, false);
}
// gfx1250 async global->LDS copy: 32B per lane (two B128 ops; INST_OFFSET is
// applied to both the LDS and global address, so one address pair suffices).
__device__ __forceinline__ void async_copy32(const bf16* gsrc, unsigned ldsoff) {
    unsigned long long g = (unsigned long long)gsrc;
    asm volatile("global_load_async_to_lds_b128 %0, %1, off\n\t"
                 "global_load_async_to_lds_b128 %0, %1, off offset:16"
                 :: "v"(ldsoff), "v"(g) : "memory");
}
__device__ __forceinline__ void wait_async0() {
    asm volatile("s_wait_asynccnt 0" ::: "memory");
}

// ---------------------------------------------------------------------------
// Stage 0: pre-convert a 1024x1024 weight matrix to bf16 (one-time, tiny)
// ---------------------------------------------------------------------------
__global__ void cvt_w(const float* __restrict__ W, bf16* __restrict__ Wb)
{
    int i = (blockIdx.x * 256 + threadIdx.x) * 8;     // DM_*DM_ / 8 threads
    *reinterpret_cast<bf16x8*>(Wb + i) = cvt8(W + i);
}

// ---------------------------------------------------------------------------
// Stage 1: X[32768,1024] @ W^T[1024,1024] + bias, bf16 out.
// Block = 8 waves, tile 256(M) x 128(N); wave = 32 rows x 128 cols
// (2 A-fragments x 8 N-tiles = 16 independent WMMAs per k-step).
// W (pre-converted bf16) staged via async-to-LDS, double-buffered, one
// barrier per k-step. sched_barrier(0) fences force load-clause / WMMA-chain
// grouping so only one LDS-latency window is exposed per k-step.
// transposeOut=0 -> out[b][h][l][d]; transposeOut=1 -> out[b][h][d][l]
// ---------------------------------------------------------------------------
__global__ __launch_bounds__(256) void proj_gemm(const float* __restrict__ X,
                                                 const bf16* __restrict__ Wb,
                                                 const float* __restrict__ bias,
                                                 bf16* __restrict__ out,
                                                 int transposeOut)
{
    __shared__ bf16 sW[2][128 * 40];
    const int tid  = threadIdx.x;
    const int wave = tid >> 5, lane = tid & 31;
    const int half = lane >> 4, l15 = lane & 15;
    const int Mbase = blockIdx.x * 256;
    const int Nbase = blockIdx.y * 128;
    const int rowBase = Mbase + wave * 32;

    // staging assignment: thread -> (row sn, 16-col chunk skh) of the W tile
    const int sn  = tid >> 1;
    const int skh = (tid & 1) * 16;
    const bf16* wsrc = Wb + (size_t)(Nbase + sn) * DM_ + skh;
    const unsigned ldsA = (unsigned)(uintptr_t)&sW[0][sn * 40 + skh];
    const unsigned ldsB = (unsigned)(uintptr_t)&sW[1][sn * 40 + skh];

    v8f acc[2][8];
#pragma unroll
    for (int mf = 0; mf < 2; mf++)
#pragma unroll
        for (int i = 0; i < 8; i++) acc[mf][i] = zero8();

    const float* xrow0 = X + (size_t)(rowBase + l15) * DM_;
    const float* xrow1 = X + (size_t)(rowBase + 16 + l15) * DM_;

    async_copy32(wsrc, ldsA);              // tile 0 -> buf 0
    wait_async0();
    __syncthreads();

    for (int ks = 0; ks < 32; ++ks) {
        const int kbase = ks * 32;
        const int cur = ks & 1;
        if (ks < 31)                        // prefetch tile ks+1 -> other buf
            async_copy32(wsrc + kbase + 32, cur ? ldsA : ldsB);

        const float* p0 = xrow0 + kbase + half * 8;
        const float* p1 = xrow1 + kbase + half * 8;
        __builtin_prefetch(p0 + 128, 0, 0);                 // global_prefetch_b8
        __builtin_prefetch(p1 + 128, 0, 0);
        // ---- raw A loads first (one clause, 8 loads in flight) ----
        float4 a0 = *reinterpret_cast<const float4*>(p0);
        float4 a1 = *reinterpret_cast<const float4*>(p0 + 4);
        float4 a2 = *reinterpret_cast<const float4*>(p0 + 16);
        float4 a3 = *reinterpret_cast<const float4*>(p0 + 20);
        float4 b0 = *reinterpret_cast<const float4*>(p1);
        float4 b1 = *reinterpret_cast<const float4*>(p1 + 4);
        float4 b2 = *reinterpret_cast<const float4*>(p1 + 16);
        float4 b3 = *reinterpret_cast<const float4*>(p1 + 20);
        __builtin_amdgcn_sched_barrier(0);
        v16bf af0 = cvt16(a0, a1, a2, a3);
        v16bf af1 = cvt16(b0, b1, b2, b3);

        // ---- all B-fragment LDS loads, then the 16-WMMA chain ----
        bf16x8 blo[8], bhi[8];
#pragma unroll
        for (int nt = 0; nt < 8; ++nt) {
            const bf16* bp = &sW[cur][(nt * 16 + l15) * 40 + half * 8];
            blo[nt] = *reinterpret_cast<const bf16x8*>(bp);
            bhi[nt] = *reinterpret_cast<const bf16x8*>(bp + 16);
        }
        __builtin_amdgcn_sched_barrier(0);
#pragma unroll
        for (int nt = 0; nt < 8; ++nt) {
            v16bf bfrag = join16(blo[nt], bhi[nt]);
            acc[0][nt] = wmma_bf16(af0, bfrag, acc[0][nt]);
            acc[1][nt] = wmma_bf16(af1, bfrag, acc[1][nt]);
        }

        wait_async0();
        __syncthreads();
    }

#pragma unroll
    for (int mf = 0; mf < 2; mf++) {
        const int mrowBase = rowBase + mf * 16;
#pragma unroll
        for (int nt = 0; nt < 8; ++nt) {
            const int col = Nbase + nt * 16 + l15;
            const float bv = bias[col];
            const int h = col >> 6, d = col & 63;
            if (!transposeOut) {
#pragma unroll
                for (int i = 0; i < 8; i++) {
                    const int m = mrowBase + i + half * 8;
                    const int b = m >> 12, l = m & 4095;
                    out[((size_t)(b * H_ + h) * L_ + l) * DH_ + d] = (bf16)(acc[mf][nt][i] + bv);
                }
            } else {
                const int m0 = mrowBase + half * 8;
                const int b = m0 >> 12, l0 = m0 & 4095;
                bf16x8 pk;
#pragma unroll
                for (int i = 0; i < 8; i++) pk[i] = (bf16)(acc[mf][nt][i] + bv);
                *reinterpret_cast<bf16x8*>(
                    &out[((size_t)(b * H_ + h) * DH_ + d) * L_ + l0]) = pk;
            }
        }
    }
}

// ---------------------------------------------------------------------------
// Stage 2a: gather sampled keys (padded to 48 rows, zeros past U)
// ---------------------------------------------------------------------------
__global__ void gather_ksamp(const bf16* __restrict__ kmat,
                             const int* __restrict__ idx,
                             bf16* __restrict__ ksamp)
{
    int e = blockIdx.x * blockDim.x + threadIdx.x;       // B*H*48*64
    if (e >= B_ * H_ * UP_ * DH_) return;
    int d  = e & 63;
    int u  = (e >> 6) % UP_;
    int bh = e / (UP_ * DH_);
    bf16 v = (bf16)0.0f;
    if (u < U_) {
        int l = idx[u];
        v = kmat[((size_t)bh * L_ + l) * DH_ + d];
    }
    ksamp[e] = v;
}

// ---------------------------------------------------------------------------
// Stage 2b: M[l] = max_s(q.k_samp) - mean_s(q.k_samp) via WMMA.
// ---------------------------------------------------------------------------
__global__ __launch_bounds__(256) void qk_metric(const bf16* __restrict__ q,
                                                 const bf16* __restrict__ ksamp,
                                                 float* __restrict__ Mmet)
{
    const int tid  = threadIdx.x;
    const int wave = tid >> 5, lane = tid & 31;
    const int half = lane >> 4, l15 = lane & 15;
    const int tile = blockIdx.x * 8 + wave;              // 32768 tiles
    const int bh = tile >> 8;
    const int lt = tile & 255;

    const bf16* qb = q + (size_t)bh * L_ * DH_;
    const bf16* kb = ksamp + (size_t)bh * UP_ * DH_;

    bf16x8 alo[2], ahi[2], blo[3][2], bhi[3][2];
#pragma unroll
    for (int ksi = 0; ksi < 2; ++ksi) {
        const bf16* ap = qb + (lt * 16 + l15) * DH_ + ksi * 32 + half * 8;
        alo[ksi] = *reinterpret_cast<const bf16x8*>(ap);
        ahi[ksi] = *reinterpret_cast<const bf16x8*>(ap + 16);
    }
#pragma unroll
    for (int nt = 0; nt < 3; ++nt)
#pragma unroll
        for (int ksi = 0; ksi < 2; ++ksi) {
            const bf16* bp = kb + (nt * 16 + l15) * DH_ + ksi * 32 + half * 8;
            blo[nt][ksi] = *reinterpret_cast<const bf16x8*>(bp);
            bhi[nt][ksi] = *reinterpret_cast<const bf16x8*>(bp + 16);
        }
    __builtin_amdgcn_sched_barrier(0);
    v8f acc[3];
#pragma unroll
    for (int nt = 0; nt < 3; ++nt) {
        acc[nt] = zero8();
#pragma unroll
        for (int ksi = 0; ksi < 2; ++ksi)
            acc[nt] = wmma_bf16(join16(alo[ksi], ahi[ksi]),
                                join16(blo[nt][ksi], bhi[nt][ksi]), acc[nt]);
    }
#pragma unroll
    for (int i = 0; i < 8; i++) {
        float mx = -3.0e38f, sm = 0.0f;
#pragma unroll
        for (int nt = 0; nt < 3; ++nt) {
            int cg = nt * 16 + l15;
            float v = acc[nt][i];
            bool valid = cg < U_;
            mx = fmaxf(mx, valid ? v : -3.0e38f);
            sm += valid ? v : 0.0f;
        }
#pragma unroll
        for (int s = 1; s < 16; s <<= 1) {
            mx = fmaxf(mx, __shfl_xor(mx, s, 32));
            sm += __shfl_xor(sm, s, 32);
        }
        if (l15 == i)
            Mmet[(size_t)bh * L_ + lt * 16 + i + half * 8] = mx - sm * (1.0f / U_);
    }
}

// ---------------------------------------------------------------------------
// Stage 3: top-45 argmax selection per (b,h) + gather q_red (padded, zeros)
// ---------------------------------------------------------------------------
__global__ __launch_bounds__(256) void topk_gather(const float* __restrict__ Mmet,
                                                   const bf16* __restrict__ q,
                                                   bf16* __restrict__ qred)
{
    __shared__ float vals[L_];
    __shared__ float rv[256];
    __shared__ int   ri[256];
    __shared__ int   sel[U_];
    const int bh  = blockIdx.x;
    const int tid = threadIdx.x;
    for (int j = tid; j < L_; j += 256) vals[j] = Mmet[(size_t)bh * L_ + j];
    __syncthreads();
    for (int it = 0; it < U_; ++it) {
        float best = -3.0e38f; int bi = 0;
        for (int j = tid; j < L_; j += 256) {
            float v = vals[j];
            if (v > best) { best = v; bi = j; }
        }
        rv[tid] = best; ri[tid] = bi;
        __syncthreads();
        for (int s = 128; s > 0; s >>= 1) {
            if (tid < s && rv[tid + s] > rv[tid]) { rv[tid] = rv[tid + s]; ri[tid] = ri[tid + s]; }
            __syncthreads();
        }
        if (tid == 0) { sel[it] = ri[0]; vals[ri[0]] = -3.0e38f; }
        __syncthreads();
    }
    for (int e = tid; e < UP_ * DH_; e += 256) {
        int u = e >> 6, d = e & 63;
        bf16 v = (bf16)0.0f;
        if (u < U_) v = q[((size_t)bh * L_ + sel[u]) * DH_ + d];
        qred[(size_t)bh * UP_ * DH_ + e] = v;
    }
}

// ---------------------------------------------------------------------------
// Stage 4: flash attention: 48 queries x 4096 keys per (b,h).
// 8 waves each own 512 keys; k/v fragments hoisted (shared across the 3
// M-tiles); online softmax; P transposed through LDS; partials merged in LDS.
// ---------------------------------------------------------------------------
__global__ __launch_bounds__(256) void flash_sparse(const bf16* __restrict__ qred,
                                                    const bf16* __restrict__ kmat,
                                                    const bf16* __restrict__ vt,
                                                    float* __restrict__ ctx)
{
    __shared__ float sBuf[8 * UP_ * DH_];   // per-wave: P tile (bf16) then raw ctx
    __shared__ float sM[8 * UP_], sS[8 * UP_];
    const int bh   = blockIdx.x;
    const int tid  = threadIdx.x;
    const int wave = tid >> 5, lane = tid & 31;
    const int half = lane >> 4, l15 = lane & 15;

    const bf16* qb = qred + (size_t)bh * UP_ * DH_;
    const bf16* kb = kmat + (size_t)bh * L_ * DH_;
    const bf16* vb = vt   + (size_t)bh * DH_ * L_;

    bf16*  Pw = reinterpret_cast<bf16*>(&sBuf[wave * UP_ * DH_]); // 48 x stride40 bf16
    float* Cw = &sBuf[wave * UP_ * DH_];

    v16bf qf[3][2];
#pragma unroll
    for (int t = 0; t < 3; t++)
#pragma unroll
        for (int ksi = 0; ksi < 2; ksi++) {
            const bf16* ap = qb + (t * 16 + l15) * DH_ + ksi * 32 + half * 8;
            qf[t][ksi] = join16(*reinterpret_cast<const bf16x8*>(ap),
                                *reinterpret_cast<const bf16x8*>(ap + 16));
        }

    v8f cacc[3][4];
    float mrow[3][8], srow[3][8];
#pragma unroll
    for (int t = 0; t < 3; t++) {
#pragma unroll
        for (int nd = 0; nd < 4; nd++) cacc[t][nd] = zero8();
#pragma unroll
        for (int i = 0; i < 8; i++) { mrow[t][i] = -1.0e30f; srow[t][i] = 0.0f; }
    }

    const int kstart = wave * (L_ / 8);
    for (int c = 0; c < 16; c++) {
        const int kbase = kstart + c * 32;
        // hoisted k/v fragments (same for all 3 M-tiles), loads clustered
        bf16x8 klo[2][2], khi[2][2], vlo[4], vhi[4];
#pragma unroll
        for (int nt = 0; nt < 2; nt++)
#pragma unroll
            for (int ksi = 0; ksi < 2; ksi++) {
                const bf16* bp = kb + (size_t)(kbase + nt * 16 + l15) * DH_ + ksi * 32 + half * 8;
                klo[nt][ksi] = *reinterpret_cast<const bf16x8*>(bp);
                khi[nt][ksi] = *reinterpret_cast<const bf16x8*>(bp + 16);
            }
#pragma unroll
        for (int nd = 0; nd < 4; nd++) {
            const bf16* bp = vb + (size_t)(nd * 16 + l15) * L_ + kbase + half * 8;
            vlo[nd] = *reinterpret_cast<const bf16x8*>(bp);
            vhi[nd] = *reinterpret_cast<const bf16x8*>(bp + 16);
        }
        __builtin_amdgcn_sched_barrier(0);

#pragma unroll
        for (int t = 0; t < 3; t++) {
            v8f sacc[2];
#pragma unroll
            for (int nt = 0; nt < 2; nt++) {
                sacc[nt] = zero8();
#pragma unroll
                for (int ksi = 0; ksi < 2; ksi++)
                    sacc[nt] = wmma_bf16(qf[t][ksi],
                                         join16(klo[nt][ksi], khi[nt][ksi]), sacc[nt]);
#pragma unroll
                for (int i = 0; i < 8; i++) sacc[nt][i] *= SCALE_;
            }
            float cm8[8];
#pragma unroll
            for (int i = 0; i < 8; i++) {
                float m2 = fmaxf(sacc[0][i], sacc[1][i]);
#pragma unroll
                for (int s = 1; s < 16; s <<= 1) m2 = fmaxf(m2, __shfl_xor(m2, s, 32));
                cm8[i] = m2;
            }
#pragma unroll
            for (int i = 0; i < 8; i++) {
                float nm = fmaxf(mrow[t][i], cm8[i]);
                float f  = __expf(mrow[t][i] - nm);
                float p0 = __expf(sacc[0][i] - nm);
                float p1 = __expf(sacc[1][i] - nm);
                float cs = p0 + p1;
#pragma unroll
                for (int s = 1; s < 16; s <<= 1) cs += __shfl_xor(cs, s, 32);
                srow[t][i] = srow[t][i] * f + cs;
                mrow[t][i] = nm;
#pragma unroll
                for (int nd = 0; nd < 4; nd++) cacc[t][nd][i] *= f;
                int row = t * 16 + i + half * 8;
                Pw[row * 40 + l15]      = (bf16)p0;
                Pw[row * 40 + 16 + l15] = (bf16)p1;
            }
        }
        // ctx += P(48x32) @ v(32x64)
#pragma unroll
        for (int t = 0; t < 3; t++) {
            const bf16* ap = &Pw[(t * 16 + l15) * 40 + half * 8];
            v16bf pf = join16(*reinterpret_cast<const bf16x8*>(ap),
                              *reinterpret_cast<const bf16x8*>(ap + 16));
#pragma unroll
            for (int nd = 0; nd < 4; nd++)
                cacc[t][nd] = wmma_bf16(pf, join16(vlo[nd], vhi[nd]), cacc[t][nd]);
        }
    }

    // dump per-wave partials
#pragma unroll
    for (int t = 0; t < 3; t++)
#pragma unroll
        for (int i = 0; i < 8; i++) {
            int row = t * 16 + i + half * 8;
            if (l15 == i) {
                sM[wave * UP_ + row] = mrow[t][i];
                sS[wave * UP_ + row] = srow[t][i];
            }
        }
#pragma unroll
    for (int t = 0; t < 3; t++)
#pragma unroll
        for (int nd = 0; nd < 4; nd++)
#pragma unroll
            for (int i = 0; i < 8; i++) {
                int row = t * 16 + i + half * 8;
                Cw[row * DH_ + nd * 16 + l15] = cacc[t][nd][i];
            }
    __syncthreads();
    // merge 8 waves + normalize
    for (int e = tid; e < UP_ * DH_; e += 256) {
        int r = e >> 6;
        float gm = -3.0e38f;
#pragma unroll
        for (int w = 0; w < 8; w++) gm = fmaxf(gm, sM[w * UP_ + r]);
        float gs = 0.0f, val = 0.0f;
#pragma unroll
        for (int w = 0; w < 8; w++) {
            float f = __expf(sM[w * UP_ + r] - gm);
            gs  += f * sS[w * UP_ + r];
            val += f * sBuf[w * UP_ * DH_ + e];
        }
        ctx[(size_t)bh * UP_ * DH_ + e] = val / gs;
    }
}

// ---------------------------------------------------------------------------
// Stage 5: mean over u (<45) with the reference's [B, d*H + h] flattening
// ---------------------------------------------------------------------------
__global__ void ctx_mean_k(const float* __restrict__ ctx, float* __restrict__ cm)
{
    int id = blockIdx.x * blockDim.x + threadIdx.x;  // B*1024
    if (id >= B_ * DM_) return;
    int b = id >> 10;
    int hd = id & 1023;
    int h = hd >> 6, d = hd & 63;
    const float* p = ctx + (size_t)(b * H_ + h) * UP_ * DH_ + d;
    float s = 0.0f;
    for (int u = 0; u < U_; ++u) s += p[u * DH_];
    cm[b * DM_ + d * H_ + h] = s * (1.0f / U_);
}

// ---------------------------------------------------------------------------
// Stage 6: out = cm @ Wo^T + bo   (tiny; one wave per output element)
// ---------------------------------------------------------------------------
__global__ __launch_bounds__(256) void final_proj(const float* __restrict__ cm,
                                                  const float* __restrict__ Wo,
                                                  const float* __restrict__ bo,
                                                  float* __restrict__ out)
{
    const int tid = threadIdx.x;
    const int wave = tid >> 5, lane = tid & 31;
    const int ow = blockIdx.x * 8 + wave;            // B*1024
    const int b = ow >> 10, n = ow & 1023;
    float acc = 0.0f;
    for (int k = lane; k < DM_; k += 32)
        acc += cm[b * DM_ + k] * Wo[(size_t)n * DM_ + k];
#pragma unroll
    for (int s = 1; s < 32; s <<= 1) acc += __shfl_xor(acc, s, 32);
    if (lane == 0) out[ow] = acc + bo[n];
}

extern "C" void kernel_launch(void* const* d_in, const int* in_sizes, int n_in,
                              void* d_out, int out_size, void* d_ws, size_t ws_size,
                              hipStream_t stream)
{
    const float* Q  = (const float*)d_in[0];
    const float* K  = (const float*)d_in[1];
    const float* V  = (const float*)d_in[2];
    const float* Wq = (const float*)d_in[3];
    const float* bq = (const float*)d_in[4];
    const float* Wk = (const float*)d_in[5];
    const float* bk = (const float*)d_in[6];
    const float* Wv = (const float*)d_in[7];
    const float* bv = (const float*)d_in[8];
    const float* Wo = (const float*)d_in[9];
    const float* bo = (const float*)d_in[10];
    const int*  idx = (const int*)d_in[11];
    float* out = (float*)d_out;

    const size_t QKV = (size_t)B_ * H_ * L_ * DH_;        // 33,554,432
    const size_t RED = (size_t)B_ * H_ * UP_ * DH_;       // 393,216
    bf16*  qb    = (bf16*)d_ws;
    bf16*  kb    = qb + QKV;
    bf16*  vt    = kb + QKV;
    bf16*  ksamp = vt + QKV;
    bf16*  qred  = ksamp + RED;
    float* Mmet  = (float*)(qred + RED);
    float* ctx   = Mmet + (size_t)B_ * H_ * L_;
    float* cm    = ctx + RED;
    bf16*  wqb   = (bf16*)(cm + (size_t)B_ * DM_);
    bf16*  wkb   = wqb + (size_t)DM_ * DM_;
    bf16*  wvb   = wkb + (size_t)DM_ * DM_;

    cvt_w<<<(DM_ * DM_) / (256 * 8), 256, 0, stream>>>(Wq, wqb);
    cvt_w<<<(DM_ * DM_) / (256 * 8), 256, 0, stream>>>(Wk, wkb);
    cvt_w<<<(DM_ * DM_) / (256 * 8), 256, 0, stream>>>(Wv, wvb);

    dim3 pg(128, 8);   // 256 M-rows per block, 128 N-cols
    proj_gemm<<<pg, 256, 0, stream>>>(Q, wqb, bq, qb, 0);
    proj_gemm<<<pg, 256, 0, stream>>>(K, wkb, bk, kb, 0);
    proj_gemm<<<pg, 256, 0, stream>>>(V, wvb, bv, vt, 1);
    gather_ksamp<<<(B_ * H_ * UP_ * DH_) / 256, 256, 0, stream>>>(kb, idx, ksamp);
    qk_metric<<<(B_ * H_ * (L_ / 16)) / 8, 256, 0, stream>>>(qb, ksamp, Mmet);
    topk_gather<<<B_ * H_, 256, 0, stream>>>(Mmet, qb, qred);
    flash_sparse<<<B_ * H_, 256, 0, stream>>>(qred, kb, vt, ctx);
    ctx_mean_k<<<(B_ * DM_) / 256, 256, 0, stream>>>(ctx, cm);
    final_proj<<<(B_ * DM_) / 8, 256, 0, stream>>>(cm, Wo, bo, out);
}
